// AttentionLePE_60730837565680
// MI455X (gfx1250) — matrix operand order repaired and checked by reading, not compiled, not run on hardware
//
#include <hip/hip_runtime.h>

typedef __attribute__((ext_vector_type(16))) _Float16 v16h;
typedef __attribute__((ext_vector_type(8)))  _Float16 v8h;
typedef __attribute__((ext_vector_type(8)))  float    v8f;
typedef __attribute__((ext_vector_type(2)))  float    v2f;
typedef __attribute__((ext_vector_type(8)))  unsigned v8u;

#define BATCH 2
#define CCH   128
#define HH    56
#define NTOT  3136      // 56*56
#define NHEAD 4
#define HD    32
#define NT_PER_B 196    // NTOT/16

// ---------------- helpers ----------------

// Assemble a 16-half A/B operand from two 16-byte chunks (per-lane).
__device__ __forceinline__ v16h load16(const _Float16* lo, const _Float16* hi) {
    v8h a = *(const v8h*)lo;
    v8h b = *(const v8h*)hi;
    v16h r;
#pragma unroll
    for (int i = 0; i < 8; ++i) { r[i] = a[i]; r[8 + i] = b[i]; }
    return r;
}

__device__ __forceinline__ v8f wmma_f16(v16h a, v16h b, v8f c) {
    return __builtin_amdgcn_wmma_f32_16x16x32_f16(false, a, false, b, (short)0, c, false, false);
}

__device__ __forceinline__ unsigned pk16(float a, float b) {
    auto h = __builtin_amdgcn_cvt_pkrtz(a, b);   // __fp16 ext_vector(2)
    return __builtin_bit_cast(unsigned, h);
}

// ---------------- kernel 1: qkv 1x1 conv (fp32 WMMA GEMM) ----------------
// qkv[o, b, n] = sum_c W[o,c] x[b,c,n] + bias[o]; scatter into q/k/v layouts.
// One wave computes a 16(o) x 16(n) tile; K-loop over c in steps of 4.
__global__ void qkv_kernel(const float* __restrict__ x, const float* __restrict__ w,
                           const float* __restrict__ bias,
                           _Float16* __restrict__ qh, _Float16* __restrict__ kh,
                           _Float16* __restrict__ vt, float* __restrict__ vf) {
    const int lane = threadIdx.x & 31;
    const int l16  = lane & 15;
    const int half = lane >> 4;
    const int nt   = blockIdx.x;            // 0..391
    const int b    = nt / NT_PER_B;
    const int n0   = (nt % NT_PER_B) * 16;
    const int o0   = blockIdx.y * 16;       // 0..368

    const float* xb = x + (size_t)b * CCH * NTOT;
    v8f acc = {};
#pragma unroll 4
    for (int c = 0; c < CCH; c += 4) {
        // A = W tile 16x4: lane row o=l16, VGPR v holds K = 2*half + v
        v2f a = *(const v2f*)(w + (size_t)(o0 + l16) * CCH + c + 2 * half);
        // B = X tile 4x16: lane col n=l16, VGPR v holds K = 2*half + v
        v2f bb;
        bb[0] = xb[(size_t)(c + 2 * half)     * NTOT + n0 + l16];
        bb[1] = xb[(size_t)(c + 2 * half + 1) * NTOT + n0 + l16];
        acc = __builtin_amdgcn_wmma_f32_16x16x4_f32(false, a, false, bb, (short)0, acc, false, false);
    }
    const float scale = 0.1767766952966369f; // hd^-0.5, folded into q
#pragma unroll
    for (int v = 0; v < 8; ++v) {
        const int o = o0 + v + 8 * half;     // D row
        const int n = n0 + l16;              // D col
        const float val = acc[v] + bias[o];
        if (o < CCH) {                       // q -> f16 [b][h][n][hd], pre-scaled
            const int hh = o >> 5, d = o & 31;
            qh[((size_t)(b * NHEAD + hh) * NTOT + n) * HD + d] = (_Float16)(val * scale);
        } else if (o < 2 * CCH) {            // k -> f16 [b][h][n][hd]
            const int cc = o - CCH, hh = cc >> 5, d = cc & 31;
            kh[((size_t)(b * NHEAD + hh) * NTOT + n) * HD + d] = (_Float16)val;
        } else {                             // v -> f16 transposed [b][h][hd][n] + f32 copy
            const int cc = o - 2 * CCH, hh = cc >> 5, d = cc & 31;
            vt[((size_t)(b * NHEAD + hh) * HD + d) * NTOT + n] = (_Float16)val;
            vf[((size_t)b * CCH + cc) * NTOT + n] = val;
        }
    }
}

// ---------------- kernel 2: flash attention (f16 WMMA) ----------------
// Each wave owns a 16-query block of one (b,head); streams keys 32 at a time.
// S^T formulation keeps the query index on lanes => softmax stats are per-lane.
__global__ void attn_kernel(const _Float16* __restrict__ qh, const _Float16* __restrict__ kh,
                            const _Float16* __restrict__ vt, float* __restrict__ attn_out) {
    const int lane = threadIdx.x & 31;
    const int wave = threadIdx.x >> 5;
    const int l16  = lane & 15;
    const int half = lane >> 4;
    const int bh   = blockIdx.x / 49;                 // 0..7 = b*4+h
    const int qb   = (blockIdx.x % 49) * 4 + wave;    // 0..195

    const _Float16* qbase = qh + (size_t)bh * NTOT * HD;
    const _Float16* kbase = kh + (size_t)bh * NTOT * HD;
    const _Float16* vbase = vt + (size_t)bh * HD * NTOT;

    // Q block as B operand (fixed): lane col n=l16, K(d) = 8*half + s (s<8), 16 + 8*half + (s-8) (s>=8)
    const _Float16* qrow = qbase + (size_t)(qb * 16 + l16) * HD;
    const v16h b_q = load16(qrow + half * 8, qrow + 16 + half * 8);

    v8f acc_a = {}, acc_b = {};            // O^T accum: cols n on lanes, rows d = v+8*half (+16)
    float m_run = -3.0e38f, l_run = 0.f;

    for (int kb = 0; kb < NTOT / 32; ++kb) {
        // --- S^T tiles: A = K rows (16 keys), K-dim = hd=32
        const _Float16* k0 = kbase + (size_t)(kb * 32 + l16) * HD;
        const _Float16* k1 = k0 + 16 * HD;
        const v16h a_k0 = load16(k0 + half * 8, k0 + 16 + half * 8);
        const v16h a_k1 = load16(k1 + half * 8, k1 + 16 + half * 8);
        v8f z = {};
        v8f s0 = wmma_f16(a_k0, b_q, z);   // rows m = v+8h   (keys kb*32+0..15)
        v8f s1 = wmma_f16(a_k1, b_q, z);   // rows m = 16+v+8h (keys kb*32+16..31)

        // --- online softmax over the 32 keys (per query column n = l16)
        float tm = s0[0];
#pragma unroll
        for (int v = 1; v < 8; ++v) tm = fmaxf(tm, s0[v]);
#pragma unroll
        for (int v = 0; v < 8; ++v) tm = fmaxf(tm, s1[v]);
        tm = fmaxf(tm, __shfl_xor(tm, 16, 32));
        const float m_new = fmaxf(m_run, tm);
        const float corr  = __expf(m_run - m_new);
        l_run *= corr;
#pragma unroll
        for (int v = 0; v < 8; ++v) { acc_a[v] *= corr; acc_b[v] *= corr; }

        float p0[8], p1[8], rs = 0.f;
#pragma unroll
        for (int v = 0; v < 8; ++v) {
            p0[v] = __expf(s0[v] - m_new);
            p1[v] = __expf(s1[v] - m_new);
            rs += p0[v] + p1[v];
        }
        rs += __shfl_xor(rs, 16, 32);
        l_run += rs;
        m_run = m_new;

        // --- build P^T as B operand: lane col n=l16, K(m) = 16*half + 2v+p
        unsigned pk0[4], pk1[4], x0[4], x1[4];
#pragma unroll
        for (int j = 0; j < 4; ++j) {
            pk0[j] = pk16(p0[2 * j], p0[2 * j + 1]);
            pk1[j] = pk16(p1[2 * j], p1[2 * j + 1]);
        }
#pragma unroll
        for (int j = 0; j < 4; ++j) {
            x0[j] = __shfl_xor(pk0[j], 16, 32);   // partner half's tile0 pairs
            x1[j] = __shfl_xor(pk1[j], 16, 32);   // partner half's tile1 pairs
        }
        v8u bu;
#pragma unroll
        for (int j = 0; j < 4; ++j) {
            bu[j]     = half ? x1[j]  : pk0[j];
            bu[4 + j] = half ? pk1[j] : x0[j];
        }
        const v16h b_p = __builtin_bit_cast(v16h, bu);

        // --- O^T += V_tile x P^T ; A = V rows (16 d), K-dim = 32 keys
        const _Float16* v0 = vbase + (size_t)l16 * NTOT + kb * 32;
        const _Float16* v1 = v0 + (size_t)16 * NTOT;
        const v16h a_v0 = load16(v0 + half * 16, v0 + half * 16 + 8);
        const v16h a_v1 = load16(v1 + half * 16, v1 + half * 16 + 8);
        acc_a = wmma_f16(a_v0, b_p, acc_a);
        acc_b = wmma_f16(a_v1, b_p, acc_b);
    }

    const float inv = 1.f / l_run;
    float* ob = attn_out + (size_t)bh * HD * NTOT + (size_t)(qb * 16 + l16);
#pragma unroll
    for (int v = 0; v < 8; ++v) {
        ob[(size_t)(v + 8 * half) * NTOT]      = acc_a[v] * inv;
        ob[(size_t)(16 + v + 8 * half) * NTOT] = acc_b[v] * inv;
    }
}

// ---------------- kernel 3: LePE 5x5 depthwise conv + residual ----------------
__global__ void lepe_kernel(const float* __restrict__ vf, const float* __restrict__ lw,
                            const float* __restrict__ lb, const float* __restrict__ attn,
                            float* __restrict__ omid) {
    const int idx = blockIdx.x * 256 + threadIdx.x;       // B*C*N exactly
    const int n = idx % NTOT;
    const int c = (idx / NTOT) % CCH;
    const int b = idx / (NTOT * CCH);
    const int y = n / HH, xcoord = n % HH;
    const float* vplane = vf + (size_t)(b * CCH + c) * NTOT;
    const float* wc = lw + c * 25;
    float acc = lb[c];
#pragma unroll
    for (int dy = -2; dy <= 2; ++dy) {
        const int yy = y + dy;
        if (yy < 0 || yy >= HH) continue;
#pragma unroll
        for (int dx = -2; dx <= 2; ++dx) {
            const int xx = xcoord + dx;
            if (xx < 0 || xx >= HH) continue;
            acc += vplane[yy * HH + xx] * wc[(dy + 2) * 5 + (dx + 2)];
        }
    }
    omid[idx] = acc + attn[idx];
}

// ---------------- kernel 4: proj 1x1 conv (fp32 WMMA GEMM) ----------------
__global__ void proj_kernel(const float* __restrict__ omid, const float* __restrict__ w,
                            const float* __restrict__ bias, float* __restrict__ out) {
    const int lane = threadIdx.x & 31;
    const int l16  = lane & 15;
    const int half = lane >> 4;
    const int nt   = blockIdx.x;
    const int b    = nt / NT_PER_B;
    const int n0   = (nt % NT_PER_B) * 16;
    const int o0   = blockIdx.y * 16;

    const float* xb = omid + (size_t)b * CCH * NTOT;
    v8f acc = {};
#pragma unroll 4
    for (int c = 0; c < CCH; c += 4) {
        v2f a = *(const v2f*)(w + (size_t)(o0 + l16) * CCH + c + 2 * half);
        v2f bb;
        bb[0] = xb[(size_t)(c + 2 * half)     * NTOT + n0 + l16];
        bb[1] = xb[(size_t)(c + 2 * half + 1) * NTOT + n0 + l16];
        acc = __builtin_amdgcn_wmma_f32_16x16x4_f32(false, a, false, bb, (short)0, acc, false, false);
    }
#pragma unroll
    for (int v = 0; v < 8; ++v) {
        const int o = o0 + v + 8 * half;
        const int n = n0 + l16;
        out[((size_t)b * CCH + o) * NTOT + n] = acc[v] + bias[o];
    }
}

// ---------------- launch ----------------
extern "C" void kernel_launch(void* const* d_in, const int* in_sizes, int n_in,
                              void* d_out, int out_size, void* d_ws, size_t ws_size,
                              hipStream_t stream) {
    const float* x      = (const float*)d_in[0];
    const float* qkv_w  = (const float*)d_in[1];
    const float* qkv_b  = (const float*)d_in[2];
    const float* lepe_w = (const float*)d_in[3];
    const float* lepe_b = (const float*)d_in[4];
    const float* proj_w = (const float*)d_in[5];
    const float* proj_b = (const float*)d_in[6];
    float* out = (float*)d_out;

    char* ws = (char*)d_ws;
    constexpr size_t SZ_H = (size_t)BATCH * NHEAD * NTOT * HD * sizeof(_Float16); // 1.6 MB
    constexpr size_t SZ_F = (size_t)BATCH * CCH * NTOT * sizeof(float);           // 3.2 MB
    _Float16* qh = (_Float16*)(ws);
    _Float16* kh = (_Float16*)(ws + SZ_H);
    _Float16* vt = (_Float16*)(ws + 2 * SZ_H);
    float*    vf = (float*)   (ws + 3 * SZ_H);
    float*    at = (float*)   (ws + 3 * SZ_H + SZ_F);
    float*    om = (float*)   (ws + 3 * SZ_H + 2 * SZ_F);

    qkv_kernel <<<dim3(BATCH * NT_PER_B, 24), 32, 0, stream>>>(x, qkv_w, qkv_b, qh, kh, vt, vf);
    attn_kernel<<<BATCH * NHEAD * 49, 128, 0, stream>>>(qh, kh, vt, at);
    lepe_kernel<<<(BATCH * CCH * NTOT) / 256, 256, 0, stream>>>(vf, lepe_w, lepe_b, at, om);
    proj_kernel<<<dim3(BATCH * NT_PER_B, 8), 32, 0, stream>>>(om, proj_w, proj_b, out);
}
